// SkeletalMotionInterpolator_32650341384829
// MI455X (gfx1250) — matrix operand order repaired and checked by reading, not compiled, not run on hardware
//
#include <hip/hip_runtime.h>
#include <hip/hip_bf16.h>
#include <math.h>

// ---------------------------------------------------------------------------
// MI455X (gfx1250) implementation.
//
// Roofline: edge gather/scatter dominates (~1 GB/layer, L2-resident since
// h/out are 100 MB each vs 192 MB L2). Dense GEMMs (~35 GFLOP total) run on
// v_wmma_f32_16x16x32_f16 (fp32 accumulate). Segment softmax uses fp32
// global atomics (L2 atomic units).
// ---------------------------------------------------------------------------

typedef __attribute__((ext_vector_type(16))) _Float16 v16h;
typedef __attribute__((ext_vector_type(8)))  float    v8f;

#define GAT_SLOPE 0.2f
#define ACT_SLOPE 0.01f
#define HCDIM 256   // H*C
#define NHEAD 4
#define CHEAD 64

// leaky_relu(v, s) == max(v, s*v) for 0 < s < 1: one v_mul + one v_max_num_f32.
__device__ __forceinline__ float lrelu(float v, float s) { return fmaxf(v, v * s); }

// Float atomic max via signed/unsigned integer punning (valid for IEEE754,
// buffer initialized to -inf).
__device__ __forceinline__ void atomicMaxF32(float* addr, float val) {
  if (val >= 0.f) atomicMax((int*)addr, __float_as_int(val));
  else            atomicMin((unsigned int*)addr, __float_as_uint(val));
}

// ---------------------------------------------------------------------------
// C[M,Nc] = act(A)[M,K] @ B[K,Nc] (+ bias).  K <= 512, K % 4 == 0.
// 256 threads = 8 waves; wave w computes rows [blockIdx.y*128 + 16w, +16),
// cols [blockIdx.x*16, +16).
// B panel is staged in LDS pre-swizzled into WMMA B-fragment order:
//   ldsBf[chunk][lane][j]  (j = 0..15 halves)  ->  one 32B LDS read per chunk.
// A is loaded as 4x float4 per chunk (branch-free except the single K tail).
// A rows are clamped (not zeroed): a WMMA D row depends only on the matching
// A row, and out-of-range rows are masked at the store.
// ACT: compile-time LeakyReLU(ACT_SLOPE) on the A operand (folds inter-layer
// activations into the next GEMM for free).
// ---------------------------------------------------------------------------
template <int ACT>
__global__ __launch_bounds__(256) void gat_gemm_wmma(
    const float* __restrict__ A, const float* __restrict__ B,
    const float* __restrict__ bias, float* __restrict__ C,
    int M, int K, int Nc)
{
  __shared__ __align__(32) _Float16 ldsBf[16 * 512];  // 16 KB: 16 chunks x 32 lanes x 16 halves
  const int tid  = threadIdx.x;
  const int lane = tid & 31;
  const int wv   = tid >> 5;
  const int colBase = blockIdx.x * 16;
  const int Kpad = (K + 31) & ~31;
  const int nChunks = Kpad >> 5;
  const int fullChunks = (K & ~31) >> 5;

  // Stage B panel directly in WMMA B-fragment order:
  // fragment element j of lane ln in chunk c corresponds to
//   k = c*32 + (j<8 ? (ln<16 ? 0 : 8) + j : 16 + (ln<16 ? 0 : 8) + (j-8)) ,  col = colBase + (ln & 15)
  for (int t = tid; t < nChunks * 512; t += 256) {
    int chunk = t >> 9;
    int r     = t & 511;
    int ln    = r >> 4;
    int j     = r & 15;
    int col   = colBase + (ln & 15);
    int k     = (chunk << 5) + ((j < 8) ? (((ln < 16) ? 0 : 8) + j) : (16 + ((ln < 16) ? 0 : 8) + (j - 8)));
    float v = (k < K && col < Nc) ? B[(size_t)k * Nc + col] : 0.f;
    ldsBf[t] = (_Float16)v;
  }
  __syncthreads();

  const int row0 = blockIdx.y * 128 + wv * 16;
  const int m    = row0 + (lane & 15);
  const int kloA = (lane < 16) ? 0 : 8;     // 16-bit A 16x32 VGPR layout
  const float* arow = A + (size_t)(m < M ? m : (M - 1)) * K;  // clamped, always in-bounds

  v8f acc = {0.f, 0.f, 0.f, 0.f, 0.f, 0.f, 0.f, 0.f};

  int c = 0;
  // Main loop: full 32-wide K chunks, branch-free vector loads.
  for (; c < fullChunks; ++c) {
    const float* ap = arow + (c << 5) + kloA;
    float4 f0 = *(const float4*)(ap);
    float4 f1 = *(const float4*)(ap + 4);
    float4 f2 = *(const float4*)(ap + 16);
    float4 f3 = *(const float4*)(ap + 20);
    float vals[16] = {f0.x, f0.y, f0.z, f0.w, f1.x, f1.y, f1.z, f1.w,
                      f2.x, f2.y, f2.z, f2.w, f3.x, f3.y, f3.z, f3.w};
    v16h afrag;
#pragma unroll
    for (int j = 0; j < 16; ++j) {
      float v = vals[j];
      if (ACT) v = lrelu(v, ACT_SLOPE);
      afrag[j] = (_Float16)v;
    }
    v16h bfrag = *(const v16h*)(ldsBf + (c << 9) + (lane << 4));
    acc = __builtin_amdgcn_wmma_f32_16x16x32_f16(
        false, afrag, false, bfrag, (short)0, acc, false, false);
  }
  // Tail chunk (runs at most once; K=80 -> 16, K=60 -> 28 remaining).
  for (; c < nChunks; ++c) {
    v16h afrag;
#pragma unroll
    for (int j = 0; j < 16; ++j) {
      int k = (c << 5) + ((j < 8) ? (kloA + j) : (16 + kloA + (j - 8)));
      float v = (k < K) ? arow[k] : 0.f;
      if (ACT) v = lrelu(v, ACT_SLOPE);
      afrag[j] = (_Float16)v;
    }
    v16h bfrag = *(const v16h*)(ldsBf + (c << 9) + (lane << 4));
    acc = __builtin_amdgcn_wmma_f32_16x16x32_f16(
        false, afrag, false, bfrag, (short)0, acc, false, false);
  }

  const int ncol = lane & 15;
  int col = colBase + ncol;
  if (col < Nc) {
    float bv = bias ? bias[col] : 0.f;
    int rbase = row0 + ((lane < 16) ? 0 : 8);   // C/D 16x16 f32 layout
#pragma unroll
    for (int vv = 0; vv < 8; ++vv) {
      int r = rbase + vv;
      if (r < M) C[(size_t)r * Nc + col] = acc[vv] + bv;
    }
  }
}

// al_s / al_d : one thread per (node, head), 64-wide dot products (float4 loads).
__global__ void gat_attn_logits(const float* __restrict__ H,
                                const float* __restrict__ a_src,
                                const float* __restrict__ a_dst,
                                float* __restrict__ als, float* __restrict__ ald,
                                int N)
{
  int t = blockIdx.x * blockDim.x + threadIdx.x;
  int node = t >> 2, head = t & 3;
  if (node >= N) return;
  const float4* hp = (const float4*)(H + (size_t)node * HCDIM + head * CHEAD);
  const float4* as = (const float4*)(a_src + head * CHEAD);
  const float4* ad = (const float4*)(a_dst + head * CHEAD);
  float s1 = 0.f, s2 = 0.f;
#pragma unroll
  for (int c = 0; c < CHEAD / 4; ++c) {
    float4 h4 = hp[c], a4 = as[c], d4 = ad[c];
    s1 += h4.x * a4.x + h4.y * a4.y + h4.z * a4.z + h4.w * a4.w;
    s2 += h4.x * d4.x + h4.y * d4.y + h4.z * d4.z + h4.w * d4.w;
  }
  als[t] = s1; ald[t] = s2;
}

// out[n,c] = bias[c];  m = -inf;  s = 0.
__global__ void gat_init_agg(float* __restrict__ m, float* __restrict__ s,
                             float* __restrict__ out, const float* __restrict__ bias,
                             int N)
{
  int idx = blockIdx.x * blockDim.x + threadIdx.x;
  if (idx < N * HCDIM) out[idx] = bias[idx & (HCDIM - 1)];
  if (idx < N * NHEAD) { m[idx] = -INFINITY; s[idx] = 0.f; }
}

// Pass 1: e = leaky_relu(al_s[src]+al_d[dst], 0.2); segment max via atomics.
__global__ void gat_edge_logits(const int* __restrict__ ei, int E, int Etot,
                                const float* __restrict__ als, const float* __restrict__ ald,
                                float* __restrict__ ebuf, float* __restrict__ m)
{
  int t = blockIdx.x * blockDim.x + threadIdx.x;
  int ee = t >> 2, hh = t & 3;
  if (ee >= Etot) return;
  int s, d;
  if (ee < E) { s = ei[ee]; d = ei[E + ee]; } else { s = d = ee - E; }  // self loops
  float e = als[s * NHEAD + hh] + ald[d * NHEAD + hh];
  e = lrelu(e, GAT_SLOPE);
  ebuf[t] = e;
  atomicMaxF32(&m[d * NHEAD + hh], e);
}

// Pass 2: p = exp(e - m[dst]); segment sum via atomics.
__global__ void gat_edge_exp(const int* __restrict__ ei, int E, int Etot,
                             float* __restrict__ ebuf, const float* __restrict__ m,
                             float* __restrict__ ssum)
{
  int t = blockIdx.x * blockDim.x + threadIdx.x;
  int ee = t >> 2, hh = t & 3;
  if (ee >= Etot) return;
  int d = (ee < E) ? ei[E + ee] : (ee - E);
  float p = __expf(ebuf[t] - m[d * NHEAD + hh]);
  ebuf[t] = p;
  atomicAdd(&ssum[d * NHEAD + hh], p);
}

// Pass 3: out[dst,c] += h[src,c] * p / s[dst]  (one block per edge, coalesced).
__global__ void gat_edge_aggregate(const int* __restrict__ ei, int E, int Etot,
                                   const float* __restrict__ ebuf,
                                   const float* __restrict__ ssum,
                                   const float* __restrict__ H,
                                   float* __restrict__ out)
{
  int t = blockIdx.x * blockDim.x + threadIdx.x;
  int ee = t >> 8, c = t & (HCDIM - 1);
  if (ee >= Etot) return;
  int s, d;
  if (ee < E) { s = ei[ee]; d = ei[E + ee]; } else { s = d = ee - E; }
  int hh = c >> 6;
  float alpha = ebuf[ee * NHEAD + hh] / ssum[d * NHEAD + hh];
  atomicAdd(&out[(size_t)d * HCDIM + c], H[(size_t)s * HCDIM + c] * alpha);
}

// ---------------------------------------------------------------------------

static inline void run_gemm(const float* A, const float* B, const float* bias,
                            float* C, int M, int K, int Nc, int act,
                            hipStream_t st)
{
  dim3 grid((Nc + 15) / 16, (M + 127) / 128);
  if (act) gat_gemm_wmma<1><<<grid, 256, 0, st>>>(A, B, bias, C, M, K, Nc);
  else     gat_gemm_wmma<0><<<grid, 256, 0, st>>>(A, B, bias, C, M, K, Nc);
}

extern "C" void kernel_launch(void* const* d_in, const int* in_sizes, int n_in,
                              void* d_out, int out_size, void* d_ws, size_t ws_size,
                              hipStream_t stream)
{
  const float* x        = (const float*)d_in[0];
  const int*   eidx     = (const int*)  d_in[1];
  const float* rootctx  = (const float*)d_in[2];
  const float* W[3]     = {(const float*)d_in[3], (const float*)d_in[7],  (const float*)d_in[11]};
  const float* Asrc[3]  = {(const float*)d_in[4], (const float*)d_in[8],  (const float*)d_in[12]};
  const float* Adst[3]  = {(const float*)d_in[5], (const float*)d_in[9],  (const float*)d_in[13]};
  const float* Bias[3]  = {(const float*)d_in[6], (const float*)d_in[10], (const float*)d_in[14]};
  const float* fc_w = (const float*)d_in[15];
  const float* fc_b = (const float*)d_in[16];
  const float* r_w1 = (const float*)d_in[17];
  const float* r_b1 = (const float*)d_in[18];
  const float* r_w2 = (const float*)d_in[19];
  const float* r_b2 = (const float*)d_in[20];
  const float* r_w3 = (const float*)d_in[21];
  const float* r_b3 = (const float*)d_in[22];

  const int F_IN  = 80;
  const int F_OUT = 80;
  const int N     = in_sizes[0] / F_IN;
  const int E     = in_sizes[1] / 2;
  const int Etot  = E + N;                  // PyG add_self_loops
  const int RIN   = 60;
  const int R     = in_sizes[2] / RIN;
  const int RHID  = in_sizes[18];           // r_b1 length
  const int ROUT  = in_sizes[22];           // r_b3 length

  // workspace layout (floats)
  float* w = (float*)d_ws;
  float* h_buf = w;               w += (size_t)N * HCDIM;
  float* aggA  = w;               w += (size_t)N * HCDIM;
  float* aggB  = w;               w += (size_t)N * HCDIM;
  float* als   = w;               w += (size_t)N * NHEAD;
  float* ald   = w;               w += (size_t)N * NHEAD;
  float* mbuf  = w;               w += (size_t)N * NHEAD;
  float* sbuf  = w;               w += (size_t)N * NHEAD;
  float* ebuf  = w;               w += (size_t)Etot * NHEAD;
  float* mlp1  = w;               w += (size_t)R * RHID;
  float* mlp2  = w;               w += (size_t)R * RHID;

  const int THR = 256;
  const int gAttn = (N * NHEAD + THR - 1) / THR;
  const int gInit = (N * HCDIM + THR - 1) / THR;
  const int gE4   = (Etot * NHEAD + THR - 1) / THR;
  const int gE256 = Etot;   // Etot*HCDIM / 256 threads

  const float* layer_in[3] = {x, aggA, aggB};
  const int    layer_K[3]  = {F_IN, HCDIM, HCDIM};
  const int    layer_act[3]= {0, 1, 1};
  float*       layer_out[3]= {aggA, aggB, aggA};

  for (int L = 0; L < 3; ++L) {
    run_gemm(layer_in[L], W[L], nullptr, h_buf, N, layer_K[L], HCDIM,
             layer_act[L], stream);
    gat_attn_logits<<<gAttn, THR, 0, stream>>>(h_buf, Asrc[L], Adst[L], als, ald, N);
    gat_init_agg<<<gInit, THR, 0, stream>>>(mbuf, sbuf, layer_out[L], Bias[L], N);
    gat_edge_logits<<<gE4, THR, 0, stream>>>(eidx, E, Etot, als, ald, ebuf, mbuf);
    gat_edge_exp<<<gE4, THR, 0, stream>>>(eidx, E, Etot, ebuf, mbuf, sbuf);
    gat_edge_aggregate<<<gE256, THR, 0, stream>>>(eidx, E, Etot, ebuf, sbuf, h_buf,
                                                  layer_out[L]);
  }

  // rot = h @ fc_w + fc_b  -> d_out[0 : N*F_OUT]
  float* out_rot  = (float*)d_out;
  float* out_root = out_rot + (size_t)N * F_OUT;
  run_gemm(aggA, fc_w, fc_b, out_rot, N, HCDIM, F_OUT, 0, stream);

  // root MLP (LeakyReLU folded into next GEMM's A operand)
  run_gemm(rootctx, r_w1, r_b1, mlp1, R, RIN,  RHID, 0, stream);
  run_gemm(mlp1,    r_w2, r_b2, mlp2, R, RHID, RHID, 1, stream);
  run_gemm(mlp2,    r_w3, r_b3, out_root, R, RHID, ROUT, 1, stream);

  (void)n_in; (void)out_size; (void)ws_size;
}
